// Transformer_69578470195515
// MI455X (gfx1250) — matrix operand order repaired and checked by reading, not compiled, not run on hardware
//
#include <hip/hip_runtime.h>
#include <math.h>

// ---------------------------------------------------------------------------
// GPT-style transformer forward for MI455X (gfx1250, wave32, WMMA).
// B=2 T=1024 E=768 H=12 HD=64 L=4 V=50257 FF=3072
// All GEMMs run on v_wmma_f32_16x16x32_f16 (f16 in, f32 accumulate), with
// double-buffered async global->LDS staging (GLOBAL_LOAD_ASYNC_TO_LDS_B128,
// tracked by ASYNCcnt / s_wait_asynccnt).
// ---------------------------------------------------------------------------

typedef _Float16 half8 __attribute__((ext_vector_type(8)));
typedef _Float16 v16h  __attribute__((ext_vector_type(16)));
typedef float    v8f   __attribute__((ext_vector_type(8)));

constexpr int cB  = 2;
constexpr int cT  = 1024;
constexpr int cE  = 768;
constexpr int cH  = 12;
constexpr int cHD = 64;
constexpr int cL  = 4;
constexpr int cV  = 50257;
constexpr int cFF = 3072;
constexpr int cR  = cB * cT;   // 2048 token rows
constexpr int cZ  = cB * cH;   // 24 (batch, head) pairs
constexpr float cEPS = 1e-3f;

// Wave-relative LDS byte offset of a generic pointer into __shared__ memory.
__device__ __forceinline__ unsigned lds_offset(const void* p) {
  return (unsigned)(unsigned long long)(__attribute__((address_space(3))) const char*)p;
}

// Async copy 16B global -> LDS (per lane). Tracked by ASYNCcnt.
__device__ __forceinline__ void async_load_b128(unsigned lds_off, const void* gsrc) {
  asm volatile("global_load_async_to_lds_b128 %0, %1, off"
               :: "v"(lds_off), "v"((unsigned long long)gsrc)
               : "memory");
}
__device__ __forceinline__ void wait_async0() {
  asm volatile("s_wait_asynccnt 0x0" ::: "memory");
}

// ---------------------------------------------------------------------------
// Generic WMMA GEMM:  C[z] = scale * A[z] @ B[z]^T(+bias)(+relu)(+res)
//   A  : f16 [M,K] row stride lda   (M must be a multiple of BM)
//   Bt : f16 [N,K] row stride ldb   (B stored transposed / column-major B)
//   C  : f32 or f16 [M,N] row stride ldc
// K must be a multiple of 64. N guarded only when NGUARD (LM head).
// Wave layout: WAVES_M x (8/WAVES_M) waves; each wave MT x NT 16x16 tiles.
// Batch z = z1*Z2 + z2 with split strides (handles per-(b,h) offsets).
// ---------------------------------------------------------------------------
template <int WAVES_M, int MT, int NT,
          bool OUT_F16, bool RELU, bool BIAS, bool RES, bool NGUARD>
__global__ __launch_bounds__(256)
void gemm_wmma_kernel(const _Float16* __restrict__ A, int lda,
                      const _Float16* __restrict__ Bt, int ldb,
                      void* __restrict__ Cv, int ldc,
                      const float* __restrict__ bias,
                      const float* __restrict__ res,
                      int M, int N, int K, float scale,
                      int Z2, long sA1, long sA2, long sB1, long sB2,
                      long sC1, long sC2) {
  constexpr int WAVES_N = 8 / WAVES_M;
  constexpr int BM  = WAVES_M * MT * 16;
  constexpr int BN  = WAVES_N * NT * 16;
  constexpr int BK  = 64;               // two WMMA k-steps per staged tile
  constexpr int LDK = BK + 8;           // +16B pad: alignment kept, banks spread
  constexpr int ACH = (BM * BK) / (256 * 8);  // 16B chunks per thread (A)
  constexpr int BCH = (BN * BK) / (256 * 8);  // 16B chunks per thread (B)

  __shared__ _Float16 As[2][BM][LDK];   // double-buffered
  __shared__ _Float16 Bs[2][BN][LDK];

  const int z  = blockIdx.z;
  const int z1 = z / Z2;
  const int z2 = z - z1 * Z2;
  A  += z1 * sA1 + z2 * sA2;
  Bt += z1 * sB1 + z2 * sB2;
  const long coff = z1 * sC1 + z2 * sC2;

  const int m0 = blockIdx.y * BM;
  const int n0 = blockIdx.x * BN;

  const int tid  = threadIdx.x;
  const int lane = tid & 31;
  const int w    = tid >> 5;
  const int wRow = (w % WAVES_M) * (MT * 16);
  const int wCol = (w / WAVES_M) * (NT * 16);

  // Issue async global->LDS staging for one BK tile into buffer `buf`.
  auto stage = [&](int buf, int k0) {
#pragma unroll
    for (int c = 0; c < ACH; ++c) {
      const int chunk = c * 256 + tid;          // lanes contiguous along rows
      const int r  = chunk >> 3;                // BK/8 = 8 chunks per row
      const int cc = (chunk & 7) * 8;
      async_load_b128(lds_offset(&As[buf][r][cc]),
                      A + (long)(m0 + r) * lda + k0 + cc);
    }
#pragma unroll
    for (int c = 0; c < BCH; ++c) {
      const int chunk = c * 256 + tid;
      const int r  = chunk >> 3;
      const int cc = (chunk & 7) * 8;
      if (NGUARD && (n0 + r >= N)) {            // zero-fill OOB rows (head GEMM)
        half8 zv;
#pragma unroll
        for (int q = 0; q < 8; ++q) zv[q] = (_Float16)0;
        *(half8*)&Bs[buf][r][cc] = zv;
      } else {
        async_load_b128(lds_offset(&Bs[buf][r][cc]),
                        Bt + (long)(n0 + r) * ldb + k0 + cc);
      }
    }
  };

  v8f acc[MT][NT];
#pragma unroll
  for (int i = 0; i < MT; ++i)
#pragma unroll
    for (int j = 0; j < NT; ++j)
#pragma unroll
      for (int q = 0; q < 8; ++q) acc[i][j][q] = 0.0f;

  const int kIters = K / BK;
  stage(0, 0);
  for (int kt = 0; kt < kIters; ++kt) {
    wait_async0();        // our async stores to LDS have landed
    __syncthreads();      // everyone's have landed (barrier waits DS too)
    if (kt + 1 < kIters) stage((kt + 1) & 1, (kt + 1) * BK);  // prefetch next

    const int buf = kt & 1;
    union U { v16h v; half8 h[2]; };
    const int sub = lane & 15;
    const int ak  = (lane >> 4) * 8;   // A frag: K chunks {ak, 16+ak}
    const int bk  = (lane >> 4) * 8;   // B frag: K chunks {bk, 16+bk}
#pragma unroll
    for (int ks = 0; ks < 2; ++ks) {
      const int kb = ks * 32;
      U a[MT], b[NT];
#pragma unroll
      for (int i = 0; i < MT; ++i) {
        const int row = wRow + i * 16 + sub;
        a[i].h[0] = *(const half8*)&As[buf][row][kb + ak];
        a[i].h[1] = *(const half8*)&As[buf][row][kb + 16 + ak];
      }
#pragma unroll
      for (int j = 0; j < NT; ++j) {
        const int col = wCol + j * 16 + sub;
        b[j].h[0] = *(const half8*)&Bs[buf][col][kb + bk];
        b[j].h[1] = *(const half8*)&Bs[buf][col][kb + 16 + bk];
      }
#pragma unroll
      for (int i = 0; i < MT; ++i)
#pragma unroll
        for (int j = 0; j < NT; ++j)
          acc[i][j] = __builtin_amdgcn_wmma_f32_16x16x32_f16(
              false, a[i].v, false, b[j].v, (short)0, acc[i][j], false, false);
    }
  }

  // ---- epilogue: C/D layout: col = lane&15, rows = (lane>>4)*8 + rr ----
  float*    Cf = (float*)Cv;
  _Float16* Ch = (_Float16*)Cv;
  const int rbase = (lane >> 4) * 8;
  const int csub  = lane & 15;
#pragma unroll
  for (int i = 0; i < MT; ++i) {
#pragma unroll
    for (int j = 0; j < NT; ++j) {
      const int col = n0 + wCol + j * 16 + csub;
      if (NGUARD && col >= N) continue;
      const float bv = BIAS ? bias[col] : 0.0f;
#pragma unroll
      for (int rr = 0; rr < 8; ++rr) {
        const int row = m0 + wRow + i * 16 + rbase + rr;
        float val = acc[i][j][rr] * scale + bv;
        if (RELU) val = fmaxf(val, 0.0f);
        const long idx = coff + (long)row * ldc + col;
        if (RES) val += res[idx];
        if constexpr (OUT_F16) Ch[idx] = (_Float16)val;
        else                   Cf[idx] = val;
      }
    }
  }
}

// ---------------------------------------------------------------------------
// fp32 [K,N] -> f16 [N,K]   (weight convert + transpose, tiled through LDS)
// ---------------------------------------------------------------------------
__global__ __launch_bounds__(256)
void convert_transpose_kernel(const float* __restrict__ W,
                              _Float16* __restrict__ Wt, int K, int N) {
  __shared__ float tile[32][33];
  const int kb = blockIdx.y * 32;
  const int nb = blockIdx.x * 32;
  const int tx = threadIdx.x, ty = threadIdx.y;  // (32, 8)
  for (int r = ty; r < 32; r += 8) {
    const int k = kb + r, n = nb + tx;
    tile[r][tx] = (k < K && n < N) ? W[(long)k * N + n] : 0.0f;
  }
  __syncthreads();
  for (int r = ty; r < 32; r += 8) {
    const int n = nb + r, k = kb + tx;
    if (n < N && k < K) Wt[(long)n * K + k] = (_Float16)tile[tx][r];
  }
}

// ---------------------------------------------------------------------------
// v [B,T,H*HD] f16 -> vT [B,H,HD,T] f16
// ---------------------------------------------------------------------------
__global__ __launch_bounds__(256)
void transpose_v_kernel(const _Float16* __restrict__ v,
                        _Float16* __restrict__ vT) {
  __shared__ _Float16 tile[32][33];
  const int z = blockIdx.z;            // b*H + h
  const int b = z / cH, h = z - b * cH;
  const int t0 = blockIdx.x * 32;      // along T
  const int d0 = blockIdx.y * 32;      // along HD
  const _Float16* src = v + ((long)b * cT) * cE + h * cHD;
  _Float16* dst = vT + (long)z * cHD * cT;
  const int tx = threadIdx.x, ty = threadIdx.y;  // (32,8)
  for (int r = ty; r < 32; r += 8)
    tile[r][tx] = src[(long)(t0 + r) * cE + d0 + tx];
  __syncthreads();
  for (int r = ty; r < 32; r += 8)
    dst[(long)(d0 + r) * cT + t0 + tx] = tile[tx][r];
}

// ---------------------------------------------------------------------------
// Embedding: x[row,:] = tok_emb[tokens[row],:] + pos_emb[row % T,:]
// ---------------------------------------------------------------------------
__global__ __launch_bounds__(256)
void embed_kernel(const int* __restrict__ tokens,
                  const float* __restrict__ tok_emb,
                  const float* __restrict__ pos_emb,
                  float* __restrict__ x) {
  const int row = blockIdx.x;
  const int tok = tokens[row];
  const int t   = row & (cT - 1);
  const float* te = tok_emb + (long)tok * cE;
  const float* pe = pos_emb + (long)t * cE;
  float* xr = x + (long)row * cE;
  for (int e = threadIdx.x; e < cE; e += 256) xr[e] = te[e] + pe[e];
}

// ---------------------------------------------------------------------------
// LayerNorm over E=768 (= 256 threads x 3). Writes f32 and/or f16 outputs.
// ---------------------------------------------------------------------------
__global__ __launch_bounds__(256)
void layernorm_kernel(const float* __restrict__ x,
                      const float* __restrict__ g,
                      const float* __restrict__ b,
                      float* __restrict__ out32,
                      _Float16* __restrict__ out16) {
  const int row = blockIdx.x;
  const float* xr = x + (long)row * cE;
  const int t = threadIdx.x;
  float v0 = xr[t], v1 = xr[t + 256], v2 = xr[t + 512];

  __shared__ float red[8];
  const int lane = t & 31, wid = t >> 5;

  float s = v0 + v1 + v2;
  for (int o = 16; o > 0; o >>= 1) s += __shfl_xor(s, o, 32);
  if (lane == 0) red[wid] = s;
  __syncthreads();
  float mu = 0.0f;
#pragma unroll
  for (int i = 0; i < 8; ++i) mu += red[i];
  mu *= (1.0f / (float)cE);
  __syncthreads();

  const float d0 = v0 - mu, d1 = v1 - mu, d2 = v2 - mu;
  float ss = d0 * d0 + d1 * d1 + d2 * d2;
  for (int o = 16; o > 0; o >>= 1) ss += __shfl_xor(ss, o, 32);
  if (lane == 0) red[wid] = ss;
  __syncthreads();
  float var = 0.0f;
#pragma unroll
  for (int i = 0; i < 8; ++i) var += red[i];
  var *= (1.0f / (float)cE);
  const float inv = rsqrtf(var + cEPS);

  const float y0 = d0 * inv * g[t] + b[t];
  const float y1 = d1 * inv * g[t + 256] + b[t + 256];
  const float y2 = d2 * inv * g[t + 512] + b[t + 512];
  if (out32) {
    float* o32 = out32 + (long)row * cE;
    o32[t] = y0; o32[t + 256] = y1; o32[t + 512] = y2;
  }
  if (out16) {
    _Float16* o16 = out16 + (long)row * cE;
    o16[t] = (_Float16)y0; o16[t + 256] = (_Float16)y1; o16[t + 512] = (_Float16)y2;
  }
}

// ---------------------------------------------------------------------------
// Column softmax (reference quirk: softmax over the QUERY axis i) with causal
// mask keep(i >= j). S f32 [Z,T,T] -> P f16 [Z,T,T]. Block handles 64 columns.
// ---------------------------------------------------------------------------
__global__ __launch_bounds__(256)
void softmax_col_kernel(const float* __restrict__ S, _Float16* __restrict__ P) {
  const int z = blockIdx.y;
  const int tx = threadIdx.x;             // 0..63 -> column
  const int ty = threadIdx.y;             // 0..3  -> row phase
  const int j = blockIdx.x * 64 + tx;
  const float* Sz = S + (long)z * cT * cT;
  _Float16*    Pz = P + (long)z * cT * cT;

  __shared__ float red[4][64];

  float m = -3.4e38f;
  for (int i = ty; i < cT; i += 4)
    if (i >= j) m = fmaxf(m, Sz[(long)i * cT + j]);
  red[ty][tx] = m;
  __syncthreads();
  m = fmaxf(fmaxf(red[0][tx], red[1][tx]), fmaxf(red[2][tx], red[3][tx]));
  __syncthreads();

  float sum = 0.0f;
  for (int i = ty; i < cT; i += 4)
    if (i >= j) sum += __expf(Sz[(long)i * cT + j] - m);
  red[ty][tx] = sum;
  __syncthreads();
  sum = red[0][tx] + red[1][tx] + red[2][tx] + red[3][tx];
  const float inv = 1.0f / sum;

  for (int i = ty; i < cT; i += 4) {
    float p = (i >= j) ? __expf(Sz[(long)i * cT + j] - m) * inv : 0.0f;
    Pz[(long)i * cT + j] = (_Float16)p;
  }
}

// ---------------------------------------------------------------------------
// Host orchestration
// ---------------------------------------------------------------------------
extern "C" void kernel_launch(void* const* d_in, const int* in_sizes, int n_in,
                              void* d_out, int out_size, void* d_ws, size_t ws_size,
                              hipStream_t stream) {
  (void)in_sizes; (void)n_in; (void)out_size; (void)ws_size;

  const int*   tokens  = (const int*)d_in[0];
  const float* tok_emb = (const float*)d_in[1];
  const float* pos_emb = (const float*)d_in[2];
  const float* wq      = (const float*)d_in[3];
  const float* wk      = (const float*)d_in[4];
  const float* wv      = (const float*)d_in[5];
  const float* wo      = (const float*)d_in[6];
  const float* wo_b    = (const float*)d_in[7];
  const float* ln1_g   = (const float*)d_in[8];
  const float* ln1_b   = (const float*)d_in[9];
  const float* ln2_g   = (const float*)d_in[10];
  const float* ln2_b   = (const float*)d_in[11];
  const float* ff1_w   = (const float*)d_in[12];
  const float* ff1_b   = (const float*)d_in[13];
  const float* ff2_w   = (const float*)d_in[14];
  const float* ff2_b   = (const float*)d_in[15];
  const float* lnf_g   = (const float*)d_in[16];
  const float* lnf_b   = (const float*)d_in[17];
  const float* head_w  = (const float*)d_in[18];
  const float* head_b  = (const float*)d_in[19];
  float* out = (float*)d_out;

  // ---- workspace carve-up ----
  char* ws = (char*)d_ws;
  size_t off = 0;
  auto alloc = [&](size_t bytes) -> char* {
    char* p = ws + off;
    off += (bytes + 255) & ~(size_t)255;
    return p;
  };
  _Float16* wqT[cL]; _Float16* wkT[cL]; _Float16* wvT[cL]; _Float16* woT[cL];
  _Float16* ff1T[cL]; _Float16* ff2T[cL];
  for (int l = 0; l < cL; ++l) {
    wqT[l]  = (_Float16*)alloc((size_t)cE * cE * 2);
    wkT[l]  = (_Float16*)alloc((size_t)cE * cE * 2);
    wvT[l]  = (_Float16*)alloc((size_t)cE * cE * 2);
    woT[l]  = (_Float16*)alloc((size_t)cE * cE * 2);
    ff1T[l] = (_Float16*)alloc((size_t)cFF * cE * 2);
    ff2T[l] = (_Float16*)alloc((size_t)cE * cFF * 2);
  }
  // head padded by 128 rows; B-staging rows past V are zero-filled, not loaded
  _Float16* headT = (_Float16*)alloc(((size_t)cV + 128) * cE * 2);
  float*    x     = (float*)alloc((size_t)cR * cE * 4);
  float*    h32   = (float*)alloc((size_t)cR * cE * 4);
  _Float16* h16   = (_Float16*)alloc((size_t)cR * cE * 2);
  _Float16* q16   = (_Float16*)alloc((size_t)cR * cE * 2);
  _Float16* k16   = (_Float16*)alloc((size_t)cR * cE * 2);
  _Float16* v16   = (_Float16*)alloc((size_t)cR * cE * 2);
  _Float16* vT    = (_Float16*)alloc((size_t)cZ * cHD * cT * 2);
  _Float16* att16 = (_Float16*)alloc((size_t)cR * cE * 2);
  _Float16* ff16  = (_Float16*)alloc((size_t)cR * cFF * 2);
  float*    S     = (float*)alloc((size_t)cZ * cT * cT * 4);
  _Float16* P     = (_Float16*)alloc((size_t)cZ * cT * cT * 2);

  const float attScale = 1.0f / sqrtf((float)cE);  // source scales by emb_dim

  // ---- convert + transpose all weights to f16 [N,K] ----
  auto convT = [&](const float* W, _Float16* Wt, int K, int N) {
    dim3 g((N + 31) / 32, (K + 31) / 32);
    convert_transpose_kernel<<<g, dim3(32, 8), 0, stream>>>(W, Wt, K, N);
  };
  for (int l = 0; l < cL; ++l) {
    convT(wq + (long)l * cE * cE, wqT[l], cE, cE);
    convT(wk + (long)l * cE * cE, wkT[l], cE, cE);
    convT(wv + (long)l * cE * cE, wvT[l], cE, cE);
    convT(wo + (long)l * cE * cE, woT[l], cE, cE);
    convT(ff1_w + (long)l * cE * cFF, ff1T[l], cE, cFF);
    convT(ff2_w + (long)l * cFF * cE, ff2T[l], cFF, cE);
  }
  convT(head_w, headT, cE, cV);

  // ---- embedding ----
  embed_kernel<<<cR, 256, 0, stream>>>(tokens, tok_emb, pos_emb, x);

  // ---- transformer layers ----
  for (int l = 0; l < cL; ++l) {
    // h = LN1(x) -> f16
    layernorm_kernel<<<cR, 256, 0, stream>>>(x, ln1_g + l * cE, ln1_b + l * cE,
                                             nullptr, h16);
    // q,k,v = h @ W  (M=2048,N=768,K=768), outputs f16; tile 128x128
    {
      dim3 g(cE / 128, cR / 128, 1);
      gemm_wmma_kernel<2, 4, 2, true, false, false, false, false>
          <<<g, 256, 0, stream>>>(h16, cE, wqT[l], cE, q16, cE, nullptr, nullptr,
                                  cR, cE, cE, 1.0f, 1, 0, 0, 0, 0, 0, 0);
      gemm_wmma_kernel<2, 4, 2, true, false, false, false, false>
          <<<g, 256, 0, stream>>>(h16, cE, wkT[l], cE, k16, cE, nullptr, nullptr,
                                  cR, cE, cE, 1.0f, 1, 0, 0, 0, 0, 0, 0);
      gemm_wmma_kernel<2, 4, 2, true, false, false, false, false>
          <<<g, 256, 0, stream>>>(h16, cE, wvT[l], cE, v16, cE, nullptr, nullptr,
                                  cR, cE, cE, 1.0f, 1, 0, 0, 0, 0, 0, 0);
    }
    // vT[b,h,d,t]
    transpose_v_kernel<<<dim3(cT / 32, cHD / 32, cZ), dim3(32, 8), 0, stream>>>(
        v16, vT);
    // S[z,i,j] = scale * q_i . k_j   (K=64 -> single staged k-tile)
    {
      dim3 g(cT / 128, cT / 128, cZ);
      gemm_wmma_kernel<2, 4, 2, false, false, false, false, false>
          <<<g, 256, 0, stream>>>(q16, cE, k16, cE, S, cT, nullptr, nullptr,
                                  cT, cT, cHD, attScale, cH,
                                  (long)cT * cE, (long)cHD,
                                  (long)cT * cE, (long)cHD,
                                  (long)cH * cT * cT, (long)cT * cT);
    }
    // column softmax (over query axis) with causal mask -> P f16
    softmax_col_kernel<<<dim3(cT / 64, cZ), dim3(64, 4), 0, stream>>>(S, P);
    // att[b,i,h,d] = P[z] @ V[z]  (N=64 -> narrow 128x64 tile config)
    {
      dim3 g(cHD / 64, cT / 128, cZ);
      gemm_wmma_kernel<4, 2, 2, true, false, false, false, false>
          <<<g, 256, 0, stream>>>(P, cT, vT, cT, att16, cE, nullptr, nullptr,
                                  cT, cHD, cT, 1.0f, cH,
                                  (long)cH * cT * cT, (long)cT * cT,
                                  (long)cH * cHD * cT, (long)cHD * cT,
                                  (long)cT * cE, (long)cHD);
    }
    // x = x + att @ wo + wo_b
    {
      dim3 g(cE / 128, cR / 128, 1);
      gemm_wmma_kernel<2, 4, 2, false, false, true, true, false>
          <<<g, 256, 0, stream>>>(att16, cE, woT[l], cE, x, cE,
                                  wo_b + l * cE, x,
                                  cR, cE, cE, 1.0f, 1, 0, 0, 0, 0, 0, 0);
    }
    // h = LN2(x) -> f32 (residual) + f16 (GEMM input)
    layernorm_kernel<<<cR, 256, 0, stream>>>(x, ln2_g + l * cE, ln2_b + l * cE,
                                             h32, h16);
    // ff = relu(h @ ff1 + b1) -> f16 [2048,3072]
    {
      dim3 g(cFF / 128, cR / 128, 1);
      gemm_wmma_kernel<2, 4, 2, true, true, true, false, false>
          <<<g, 256, 0, stream>>>(h16, cE, ff1T[l], cE, ff16, cFF,
                                  ff1_b + l * cFF, nullptr,
                                  cR, cFF, cE, 1.0f, 1, 0, 0, 0, 0, 0, 0);
    }
    // x = h + ff @ ff2 + b2   (residual is post-LN h, per source quirk)
    {
      dim3 g(cE / 128, cR / 128, 1);
      gemm_wmma_kernel<2, 4, 2, false, false, true, true, false>
          <<<g, 256, 0, stream>>>(ff16, cFF, ff2T[l], cFF, x, cE,
                                  ff2_b + l * cE, h32,
                                  cR, cE, cFF, 1.0f, 1, 0, 0, 0, 0, 0, 0);
    }
  }

  // ---- final LN + LM head (N=50257 -> N-guarded instantiation) ----
  layernorm_kernel<<<cR, 256, 0, stream>>>(x, lnf_g, lnf_b, nullptr, h16);
  {
    dim3 g((cV + 127) / 128, cR / 128, 1);
    gemm_wmma_kernel<2, 4, 2, false, false, true, false, true>
        <<<g, 256, 0, stream>>>(h16, cE, headT, cE, out, cV, head_b, nullptr,
                                cR, cV, cE, 1.0f, 1, 0, 0, 0, 0, 0, 0);
  }
}
